// MultiHeadAttention_15049565405879
// MI455X (gfx1250) — matrix operand order repaired and checked by reading, not compiled, not run on hardware
//
#include <hip/hip_runtime.h>

// MHA forward for MI455X (gfx1250, wave32, WMMA).
// d_in: q,k,v [8,2048,512] f32; Wq,bq,Wk,bk,Wv,bv f32.
// d_out: context [8,2048,512] f32  followed by  attn [64,2048,2048] f32.
// Workspace layout (f16): Wt (3*512*512) | Qh (64*2048*64) | Kh | Vt  ~= 52 MB.

#define S_LEN   2048
#define DM      512
#define NH      8
#define HD      64
#define BATCH   8

typedef __attribute__((ext_vector_type(16))) _Float16 v16h;
typedef __attribute__((ext_vector_type(8)))  _Float16 v8h;
typedef __attribute__((ext_vector_type(8)))  float    v8f;
typedef __attribute__((ext_vector_type(4)))  float    v4f;

union F16x16 { v16h v; v8h h[2]; };

__device__ __forceinline__ v8f wmma16(const F16x16& a, const F16x16& b, v8f c) {
  // D = A(16x32 f16) * B(32x16 f16) + C(16x16 f32)
  return __builtin_amdgcn_wmma_f32_16x16x32_f16(
      /*neg_a=*/false, a.v, /*neg_b=*/false, b.v,
      /*c_mod=*/(short)0, c, /*reuse_a=*/false, /*reuse_b=*/false);
}

// Row reductions across the 16 lanes of a half-wave (masks <16 never cross halves).
__device__ __forceinline__ float red_max16(float x) {
  x = fmaxf(x, __shfl_xor(x, 1));
  x = fmaxf(x, __shfl_xor(x, 2));
  x = fmaxf(x, __shfl_xor(x, 4));
  x = fmaxf(x, __shfl_xor(x, 8));
  return x;
}
__device__ __forceinline__ float red_sum16(float x) {
  x += __shfl_xor(x, 1);
  x += __shfl_xor(x, 2);
  x += __shfl_xor(x, 4);
  x += __shfl_xor(x, 8);
  return x;
}

// ---------------- Kernel 0: weight transpose + f16 convert: Wt[n][k] = W[k][n]
__global__ void mha_wt_kernel(const float* __restrict__ Wq,
                              const float* __restrict__ Wk,
                              const float* __restrict__ Wv,
                              _Float16* __restrict__ Wt) {
  int idx = blockIdx.x * 256 + threadIdx.x;      // 3*512*512 threads
  int mat = idx >> 18;
  int rc  = idx & 262143;
  int n   = rc >> 9;
  int kk  = rc & 511;
  const float* W = (mat == 0) ? Wq : ((mat == 1) ? Wk : Wv);
  Wt[(size_t)mat * 262144 + (size_t)n * 512 + kk] = (_Float16)W[(size_t)kk * 512 + n];
}

// ---------------- Kernel 1: fused QKV projection GEMM (f16 WMMA, f32 acc)
// One wave computes a 16x64 output tile (= one head's hd columns for 16 rows).
__global__ void mha_proj_kernel(const float* __restrict__ q,
                                const float* __restrict__ kin,
                                const float* __restrict__ vin,
                                const float* __restrict__ bq,
                                const float* __restrict__ bk,
                                const float* __restrict__ bv,
                                const _Float16* __restrict__ Wt,
                                _Float16* __restrict__ Qh,
                                _Float16* __restrict__ Kh,
                                _Float16* __restrict__ Vt) {
  int lane = threadIdx.x & 31;
  int w    = threadIdx.x >> 5;
  int wid  = blockIdx.x * 4 + w;          // 3 * 1024 * 8 waves
  int mat  = wid >> 13;                   // / 8192
  int rem  = wid & 8191;
  int rowtile = rem >> 3;                 // 0..1023 (16-row tiles of B*S)
  int h       = rem & 7;                  // head

  const float*    X    = (mat == 0) ? q  : ((mat == 1) ? kin : vin);
  const float*    bias = (mat == 0) ? bq : ((mat == 1) ? bk  : bv);
  const _Float16* W    = Wt + (size_t)mat * 262144;

  int ln   = lane & 15;
  int half = lane >> 4;
  int row  = rowtile * 16 + ln;           // A-matrix M = ln (both halves)
  const float* xrow = X + (size_t)row * DM;

  v8f acc[4] = {v8f{}, v8f{}, v8f{}, v8f{}};

  for (int kk = 0; kk < 16; ++kk) {
    int k0 = kk * 32;
    // A frag: lane holds row `row`, K = 8*half+{0..7} and 16+8*half+{0..7}
    const float* xp = xrow + k0 + 8 * half;
    v4f f0 = *(const v4f*)(xp);
    v4f f1 = *(const v4f*)(xp + 4);
    v4f f2 = *(const v4f*)(xp + 16);
    v4f f3 = *(const v4f*)(xp + 20);
    F16x16 a;
#pragma unroll
    for (int i = 0; i < 4; ++i) {
      a.v[i]      = (_Float16)f0[i];
      a.v[4 + i]  = (_Float16)f1[i];
      a.v[8 + i]  = (_Float16)f2[i];
      a.v[12 + i] = (_Float16)f3[i];
    }
    // B frags: lane holds column n, K = 8*half+{0..7} and 16+8*half+{0..7} in Wt row n
#pragma unroll
    for (int ct = 0; ct < 4; ++ct) {
      const _Float16* wp = W + (size_t)(h * 64 + ct * 16 + ln) * 512 + k0 + 8 * half;
      F16x16 b;
      b.h[0] = *(const v8h*)(wp);
      b.h[1] = *(const v8h*)(wp + 16);
      acc[ct] = wmma16(a, b, acc[ct]);
    }
  }

  // Epilogue: bias add + f16 store. D layout: acc[ct][r] = (row0+r+8*half, ct*16+ln).
  int row0 = rowtile * 16;
  int bb   = row0 >> 11;          // batch
  int s0   = row0 & 2047;         // seq offset
  int p    = h * BATCH + bb;      // head-major plane

  if (mat < 2) {
    _Float16* O = (mat == 0) ? Qh : Kh;   // [p][s][64]
#pragma unroll
    for (int ct = 0; ct < 4; ++ct) {
      float bn = bias[h * 64 + ct * 16 + ln];
#pragma unroll
      for (int r = 0; r < 8; ++r) {
        int s = s0 + r + 8 * half;
        O[((size_t)p * S_LEN + s) * 64 + ct * 16 + ln] = (_Float16)(acc[ct][r] + bn);
      }
    }
  } else {
    // V stored transposed: Vt[p][col][s] so attention B-frags are contiguous.
#pragma unroll
    for (int ct = 0; ct < 4; ++ct) {
      int col  = ct * 16 + ln;
      float bn = bias[h * 64 + col];
      v8h hv;
#pragma unroll
      for (int r = 0; r < 8; ++r) hv[r] = (_Float16)(acc[ct][r] + bn);
      *(v8h*)&Vt[((size_t)p * 64 + col) * S_LEN + s0 + 8 * half] = hv;
    }
  }
}

// ---------------- Kernel 2: attention. One wave owns a 16-query block of one (h,b).
__global__ void mha_attn_kernel(const _Float16* __restrict__ Qh,
                                const _Float16* __restrict__ Kh,
                                const _Float16* __restrict__ Vt,
                                float* __restrict__ out) {
  __shared__ __align__(16) _Float16 ldsP[4][2][256];  // per-wave double slot for P tiles

  int lane = threadIdx.x & 31;
  int w    = threadIdx.x >> 5;
  int p    = blockIdx.x >> 5;                 // 0..63 = h*BATCH + b
  int rowblk = (blockIdx.x & 31) * 4 + w;     // 0..127
  int row0   = rowblk * 16;
  int ln   = lane & 15;
  int half = lane >> 4;

  const _Float16* Qp = Qh + (size_t)p * S_LEN * 64;
  const _Float16* Kp = Kh + (size_t)p * S_LEN * 64;
  const _Float16* Vp = Vt + (size_t)p * 64 * S_LEN;
  float* attn = out + (size_t)BATCH * S_LEN * DM + (size_t)p * S_LEN * S_LEN;

  // Q A-frags (K-dim = hd = 64 -> two 16x32 frags), resident for whole kernel.
  const _Float16* qr = Qp + (size_t)(row0 + ln) * 64 + 8 * half;
  F16x16 a0, a1;
  a0.h[0] = *(const v8h*)(qr);
  a0.h[1] = *(const v8h*)(qr + 16);
  a1.h[0] = *(const v8h*)(qr + 32);
  a1.h[1] = *(const v8h*)(qr + 48);

  const float scale = 0.125f;   // 1/sqrt(64)
  float m[8], s[8];
#pragma unroll
  for (int r = 0; r < 8; ++r) { m[r] = -INFINITY; s[r] = 0.0f; }

  // ---- Pass 1: online row max / exp-sum over all 128 key tiles.
  for (int j = 0; j < 128; ++j) {
    const _Float16* kr = Kp + (size_t)(j * 16 + ln) * 64 + 8 * half;
    if (j < 127) __builtin_prefetch(kr + 1024, 0, 3);
    F16x16 b0, b1;
    b0.h[0] = *(const v8h*)(kr);
    b0.h[1] = *(const v8h*)(kr + 16);
    b1.h[0] = *(const v8h*)(kr + 32);
    b1.h[1] = *(const v8h*)(kr + 48);
    v8f c = {};
    c = wmma16(a0, b0, c);
    c = wmma16(a1, b1, c);
#pragma unroll
    for (int r = 0; r < 8; ++r) {
      float sc = c[r] * scale;
      float mt = red_max16(sc);
      float mn = fmaxf(m[r], mt);
      float pe = __expf(sc - mn);
      float ps = red_sum16(pe);
      s[r] = s[r] * __expf(m[r] - mn) + ps;
      m[r] = mn;
    }
  }
  float inv[8];
#pragma unroll
  for (int r = 0; r < 8; ++r) inv[r] = 1.0f / s[r];

  // ---- Pass 2: recompute scores (K/V are L2-hot), write normalized attn (1 GB,
  // nontemporal stream), and accumulate ctx += P(16x32) x V(32x64) per tile-pair.
  v8f ctx[4] = {v8f{}, v8f{}, v8f{}, v8f{}};
  for (int j = 0; j < 128; ++j) {
    const _Float16* kr = Kp + (size_t)(j * 16 + ln) * 64 + 8 * half;
    F16x16 b0, b1;
    b0.h[0] = *(const v8h*)(kr);
    b0.h[1] = *(const v8h*)(kr + 16);
    b1.h[0] = *(const v8h*)(kr + 32);
    b1.h[1] = *(const v8h*)(kr + 48);
    v8f c = {};
    c = wmma16(a0, b0, c);
    c = wmma16(a1, b1, c);
#pragma unroll
    for (int r = 0; r < 8; ++r) {
      float sc = c[r] * scale;
      float pv = __expf(sc - m[r]) * inv[r];
      __builtin_nontemporal_store(
          pv, attn + (size_t)(row0 + r + 8 * half) * S_LEN + j * 16 + ln);
      ldsP[w][j & 1][(r + 8 * half) * 16 + ln] = (_Float16)pv;
    }
    if (j & 1) {
      // P A-frag: M = ln; K<16 -> even slot, K>=16 -> odd slot (A layout).
      F16x16 ap;
      ap.h[0] = *(const v8h*)&ldsP[w][0][ln * 16 + 8 * half];
      ap.h[1] = *(const v8h*)&ldsP[w][1][ln * 16 + 8 * half];
      int kv0 = (j >> 1) * 32;
#pragma unroll
      for (int ct = 0; ct < 4; ++ct) {
        int col = ct * 16 + ln;
        const _Float16* vr = Vp + (size_t)col * S_LEN + kv0 + 8 * half;
        F16x16 bv;
        bv.h[0] = *(const v8h*)(vr);
        bv.h[1] = *(const v8h*)(vr + 16);
        ctx[ct] = wmma16(ap, bv, ctx[ct]);
      }
    }
  }

  // ---- Store context: out[b, s, h*64 + col]
  int hh = p >> 3, bb = p & 7;
#pragma unroll
  for (int ct = 0; ct < 4; ++ct) {
#pragma unroll
    for (int r = 0; r < 8; ++r) {
      out[(size_t)(bb * S_LEN + row0 + r + 8 * half) * DM + hh * 64 + ct * 16 + ln] =
          ctx[ct][r];
    }
  }
}

extern "C" void kernel_launch(void* const* d_in, const int* in_sizes, int n_in,
                              void* d_out, int out_size, void* d_ws, size_t ws_size,
                              hipStream_t stream) {
  const float* q  = (const float*)d_in[0];
  const float* k  = (const float*)d_in[1];
  const float* v  = (const float*)d_in[2];
  const float* Wq = (const float*)d_in[3];
  const float* bq = (const float*)d_in[4];
  const float* Wk = (const float*)d_in[5];
  const float* bk = (const float*)d_in[6];
  const float* Wv = (const float*)d_in[7];
  const float* bv = (const float*)d_in[8];

  _Float16* Wt = (_Float16*)d_ws;                 // 3*512*512      (1.5 MB)
  _Float16* Qh = Wt + (size_t)3 * 262144;         // 64*2048*64     (16.8 MB)
  _Float16* Kh = Qh + (size_t)64 * S_LEN * 64;
  _Float16* Vt = Kh + (size_t)64 * S_LEN * 64;    // total ~52 MB of ws

  // 1) transpose+convert weights: 786432 elements
  mha_wt_kernel<<<3072, 256, 0, stream>>>(Wq, Wk, Wv, Wt);
  // 2) fused QKV projection: 3 mats * 1024 rowtiles * 8 heads waves, 4 waves/block
  mha_proj_kernel<<<6144, 128, 0, stream>>>(q, k, v, bq, bk, bv, Wt, Qh, Kh, Vt);
  // 3) attention: 64 planes * 32 rowblock-groups, 4 waves/block
  mha_attn_kernel<<<2048, 128, 0, stream>>>(Qh, Kh, Vt, (float*)d_out);
}
